// KernelSelfAttention_46943992545490
// MI455X (gfx1250) — compile-verified, matrix operand order checked by reading, not run on hardware
//
#include <hip/hip_runtime.h>

#define HIDN 1024
#define SEQ  2048
#define NH   16
#define HD   64
#define NM   128

typedef __attribute__((ext_vector_type(16))) __bf16 v16bf;
typedef __attribute__((ext_vector_type(8)))  float  v8f;
typedef unsigned short u16;
typedef unsigned int   u32;

__device__ __forceinline__ u16 f2bf_bits(float f) {
  u32 u = __builtin_bit_cast(u32, f);
  u32 r = u + 0x7FFFu + ((u >> 16) & 1u);   // round-to-nearest-even
  return (u16)(r >> 16);
}
__device__ __forceinline__ float bf2f(u16 h) {
  u32 u = ((u32)h) << 16;
  return __builtin_bit_cast(float, u);
}

union FragU { v16bf v; uint4 u[2]; };

// Load a 16x32 bf16 A/B fragment for v_wmma_f32_16x16x32_bf16.
// Lane L: row/col = L%16; k = (e/8)*16 + (L/16)*8 + e%8  -> two 16B chunks.
__device__ __forceinline__ v16bf load_frag(const u16* base, int stride) {
  int lane = threadIdx.x & 31;
  const u16* p = base + (lane & 15) * stride + ((lane >> 4) << 3);
  FragU f;
  f.u[0] = *(const uint4*)(p);
  f.u[1] = *(const uint4*)(p + 16);
  return f.v;
}

__device__ __forceinline__ v8f wmma_bf16(v16bf a, v16bf b, v8f c) {
  return __builtin_amdgcn_wmma_f32_16x16x32_bf16(false, a, false, b, (short)0, c, false, false);
}

// ---------------- conversion kernels ----------------
__global__ void k_conv_f32_bf16(const float* __restrict__ src, u16* __restrict__ dst, int n) {
  int i = blockIdx.x * 256 + threadIdx.x;
  if (i < n) dst[i] = f2bf_bits(src[i]);
}

// omega (H,D,M) f32 -> omT[h][m][d] bf16
__global__ void k_omega_t(const float* __restrict__ om, u16* __restrict__ omT) {
  int i = blockIdx.x * 256 + threadIdx.x;
  if (i < NH * HD * NM) {
    int h = i / (HD * NM);
    int rem = i % (HD * NM);
    int d = rem / NM, m = rem % NM;
    omT[(h * NM + m) * HD + d] = f2bf_bits(om[i]);
  }
}

// ---------------- QKV projection: out = X @ W^T + b ----------------
// 2x2 register-blocked: each wave computes a 32x32 tile (4 WMMAs per 8 b128 loads).
// Block = 256 threads = 8 waves arranged 2x4 -> 64 rows x 128 cols per block.
// grid (HIDN/128, SEQ/64, 3)
__global__ void k_qkv(const u16* __restrict__ X,
                      const u16* __restrict__ Wq, const u16* __restrict__ Wk, const u16* __restrict__ Wv,
                      const float* __restrict__ bq, const float* __restrict__ bk, const float* __restrict__ bv,
                      u16* __restrict__ Qh, u16* __restrict__ Kh, u16* __restrict__ Vt) {
  int w = threadIdx.x >> 5, lane = threadIdx.x & 31;
  int rowbase = blockIdx.y * 64 + (w >> 2) * 32;
  int colbase = blockIdx.x * 128 + (w & 3) * 32;
  int z = blockIdx.z;
  const u16* W = (z == 0) ? Wq : (z == 1) ? Wk : Wv;
  const float* bias = (z == 0) ? bq : (z == 1) ? bk : bv;
  v8f acc00 = {}, acc01 = {}, acc10 = {}, acc11 = {};
  const u16* Ab = X + rowbase * HIDN;
  const u16* Bb = W + colbase * HIDN;
  for (int k = 0; k < HIDN; k += 32) {
    v16bf a0 = load_frag(Ab + k, HIDN);
    v16bf a1 = load_frag(Ab + 16 * HIDN + k, HIDN);
    v16bf b0 = load_frag(Bb + k, HIDN);
    v16bf b1 = load_frag(Bb + 16 * HIDN + k, HIDN);
    acc00 = wmma_bf16(a0, b0, acc00);
    acc01 = wmma_bf16(a0, b1, acc01);
    acc10 = wmma_bf16(a1, b0, acc10);
    acc11 = wmma_bf16(a1, b1, acc11);
  }
#pragma unroll
  for (int i = 0; i < 2; ++i) {
#pragma unroll
    for (int j = 0; j < 2; ++j) {
      const v8f& acc = (i == 0) ? (j == 0 ? acc00 : acc01) : (j == 0 ? acc10 : acc11);
      int col = colbase + j * 16 + (lane & 15);
      float bs = bias[col];
      int h = col >> 6, d = col & 63;
      int r0 = rowbase + i * 16 + ((lane >> 4) << 3);
#pragma unroll
      for (int e = 0; e < 8; ++e) {
        u16 hb = f2bf_bits(acc[e] + bs);
        int s = r0 + e;
        if (z == 2)      Vt[(h * HD + d) * SEQ + s] = hb;     // V transposed [h][d][s]
        else if (z == 0) Qh[(h * SEQ + s) * HD + d] = hb;
        else             Kh[(h * SEQ + s) * HD + d] = hb;
      }
    }
  }
}

// ---------------- per-(h,s) L2 normalize over D ----------------
__global__ void k_norm(const u16* __restrict__ In, u16* __restrict__ Out) {
  int idx = blockIdx.x * 256 + threadIdx.x;   // h*SEQ + s
  if (idx >= NH * SEQ) return;
  const u16* p = In + idx * HD;
  float ss = 0.f;
#pragma unroll
  for (int d = 0; d < HD; ++d) { float t = bf2f(p[d]); ss += t * t; }
  float inv = 1.0f / (sqrtf(ss) + 1e-6f);
#pragma unroll
  for (int d = 0; d < HD; ++d) Out[idx * HD + d] = f2bf_bits(bf2f(p[d]) * inv);
}

// ---------------- phi = 0.125*cos(xn @ omega + b), row-normalized over M ----------------
// grid (SEQ/16, NH, 2), block 256
__global__ void k_phi(const u16* __restrict__ Qn, const u16* __restrict__ Kn,
                      const u16* __restrict__ omT, const float* __restrict__ rffb,
                      u16* __restrict__ PhiQ, u16* __restrict__ PhiK) {
  __shared__ float tile[16][NM];
  __shared__ float invn[16];
  int w = threadIdx.x >> 5, lane = threadIdx.x & 31;
  int stile = blockIdx.x * 16;
  int h = blockIdx.y, z = blockIdx.z;
  const u16* N = z ? Kn : Qn;
  u16* Out = z ? PhiK : PhiQ;
  v8f acc = {};
  const u16* Ab = N + (h * SEQ + stile) * HD;
  const u16* Bb = omT + (h * NM + w * 16) * HD;
  for (int k = 0; k < HD; k += 32)
    acc = wmma_bf16(load_frag(Ab + k, HD), load_frag(Bb + k, HD), acc);
  int m = w * 16 + (lane & 15);
  float bph = rffb[h * NM + m];
  int r0 = (lane >> 4) << 3;
#pragma unroll
  for (int e = 0; e < 8; ++e) tile[r0 + e][m] = 0.125f * __cosf(acc[e] + bph);
  __syncthreads();
  if (threadIdx.x < 16) {
    float ss = 0.f;
    for (int c = 0; c < NM; ++c) { float t = tile[threadIdx.x][c]; ss += t * t; }
    invn[threadIdx.x] = 1.0f / (sqrtf(ss) + 1e-6f);
  }
  __syncthreads();
  for (int i = threadIdx.x; i < 16 * NM; i += 256) {
    int r = i >> 7, c = i & 127;
    Out[(h * SEQ + stile + r) * NM + c] = f2bf_bits(tile[r][c] * invn[r]);
  }
}

// ---------------- fused attention (flash-style online softmax) ----------------
// grid (SEQ/16, NH), block 256. Each wave streams 8 tiles of 32 keys.
__global__ void k_attn(const u16* __restrict__ Qh, const u16* __restrict__ Kh,
                       const u16* __restrict__ Vt,
                       const u16* __restrict__ PhiQ, const u16* __restrict__ PhiK,
                       const float* __restrict__ mask, u16* __restrict__ Ctx) {
  __shared__ __align__(16) u16 ldsP[8][16 * 32];
  __shared__ float lm[8][16], ll[8][16];
  __shared__ float lO[8][16][HD];
  int w = threadIdx.x >> 5, lane = threadIdx.x & 31;
  int h = blockIdx.y, qbase = blockIdx.x * 16;

  const u16* Qb = Qh + (h * SEQ + qbase) * HD;
  const u16* Pq = PhiQ + (h * SEQ + qbase) * NM;
  v16bf qa0 = load_frag(Qb + 0, HD),  qa1 = load_frag(Qb + 32, HD);
  v16bf pa0 = load_frag(Pq + 0, NM),  pa1 = load_frag(Pq + 32, NM);
  v16bf pa2 = load_frag(Pq + 64, NM), pa3 = load_frag(Pq + 96, NM);

  float mrow[8], lrow[8], scale[8];
  v8f O0 = {}, O1 = {}, O2 = {}, O3 = {};
#pragma unroll
  for (int e = 0; e < 8; ++e) { mrow[e] = -1e30f; lrow[e] = 0.f; }
  int r0 = (lane >> 4) << 3;

  for (int it = 0; it < 8; ++it) {
    int jbase = (it * 8 + w) * 32;           // 32 keys per iteration
    v8f sc[2];
#pragma unroll
    for (int t = 0; t < 2; ++t) {
      int kb = jbase + t * 16;
      const u16* Kb = Kh + (h * SEQ + kb) * HD;
      const u16* Pk = PhiK + (h * SEQ + kb) * NM;
      v8f dt = {};
      dt = wmma_bf16(qa0, load_frag(Kb + 0, HD), dt);
      dt = wmma_bf16(qa1, load_frag(Kb + 32, HD), dt);
      v8f ph = {};
      ph = wmma_bf16(pa0, load_frag(Pk + 0, NM), ph);
      ph = wmma_bf16(pa1, load_frag(Pk + 32, NM), ph);
      ph = wmma_bf16(pa2, load_frag(Pk + 64, NM), ph);
      ph = wmma_bf16(pa3, load_frag(Pk + 96, NM), ph);
      float mk = mask[kb + (lane & 15)] * -10000.0f;
#pragma unroll
      for (int e = 0; e < 8; ++e) sc[t][e] = (0.9f * dt[e] + 0.1f * ph[e]) * 0.125f + mk;
    }
    // online softmax: reductions stay within 16-lane halves (C-layout rows)
#pragma unroll
    for (int e = 0; e < 8; ++e) {
      float mx = fmaxf(sc[0][e], sc[1][e]);
      mx = fmaxf(mx, __shfl_xor(mx, 1));
      mx = fmaxf(mx, __shfl_xor(mx, 2));
      mx = fmaxf(mx, __shfl_xor(mx, 4));
      mx = fmaxf(mx, __shfl_xor(mx, 8));
      float mn = fmaxf(mrow[e], mx);
      scale[e] = __expf(mrow[e] - mn);
      float p0 = __expf(sc[0][e] - mn), p1 = __expf(sc[1][e] - mn);
      sc[0][e] = p0; sc[1][e] = p1;
      float rs = p0 + p1;
      rs += __shfl_xor(rs, 1);
      rs += __shfl_xor(rs, 2);
      rs += __shfl_xor(rs, 4);
      rs += __shfl_xor(rs, 8);
      lrow[e] = lrow[e] * scale[e] + rs;
      mrow[e] = mn;
    }
    // P (C-layout) -> bf16 A-fragment via per-wave LDS bounce
#pragma unroll
    for (int t = 0; t < 2; ++t)
#pragma unroll
      for (int e = 0; e < 8; ++e)
        ldsP[w][(r0 + e) * 32 + t * 16 + (lane & 15)] = f2bf_bits(sc[t][e]);
    v16bf Pa = load_frag(&ldsP[w][0], 32);
#pragma unroll
    for (int e = 0; e < 8; ++e) { O0[e] *= scale[e]; O1[e] *= scale[e]; O2[e] *= scale[e]; O3[e] *= scale[e]; }
    const u16* Vb = Vt + h * HD * SEQ + jbase;
    O0 = wmma_bf16(Pa, load_frag(Vb + 0 * 16 * SEQ, SEQ), O0);
    O1 = wmma_bf16(Pa, load_frag(Vb + 1 * 16 * SEQ, SEQ), O1);
    O2 = wmma_bf16(Pa, load_frag(Vb + 2 * 16 * SEQ, SEQ), O2);
    O3 = wmma_bf16(Pa, load_frag(Vb + 3 * 16 * SEQ, SEQ), O3);
  }

  // cross-wave merge
  if ((lane & 15) == 0)
#pragma unroll
    for (int e = 0; e < 8; ++e) { lm[w][r0 + e] = mrow[e]; ll[w][r0 + e] = lrow[e]; }
#pragma unroll
  for (int e = 0; e < 8; ++e) {
    int rr = r0 + e, cc = lane & 15;
    lO[w][rr][0 * 16 + cc] = O0[e];
    lO[w][rr][1 * 16 + cc] = O1[e];
    lO[w][rr][2 * 16 + cc] = O2[e];
    lO[w][rr][3 * 16 + cc] = O3[e];
  }
  __syncthreads();
  int row = threadIdx.x >> 4;
  int d0 = (threadIdx.x & 15) * 4;
  float gm = -1e30f;
  for (int ww = 0; ww < 8; ++ww) gm = fmaxf(gm, lm[ww][row]);
  float gl = 0.f, o[4] = {0.f, 0.f, 0.f, 0.f};
  for (int ww = 0; ww < 8; ++ww) {
    float f = __expf(lm[ww][row] - gm);
    gl += ll[ww][row] * f;
#pragma unroll
    for (int j = 0; j < 4; ++j) o[j] += lO[ww][row][d0 + j] * f;
  }
  float inv = 1.0f / gl;
  int s = qbase + row;
#pragma unroll
  for (int j = 0; j < 4; ++j)
    Ctx[s * HIDN + h * HD + d0 + j] = f2bf_bits(o[j] * inv);
}

// ---------------- output projection: out = Ctx @ Wo^T + bo (f32) ----------------
// Same 2x2 register blocking as k_qkv. grid (HIDN/128, SEQ/64)
__global__ void k_out(const u16* __restrict__ Ctx, const u16* __restrict__ Wo,
                      const float* __restrict__ bo, float* __restrict__ Out) {
  int w = threadIdx.x >> 5, lane = threadIdx.x & 31;
  int rowbase = blockIdx.y * 64 + (w >> 2) * 32;
  int colbase = blockIdx.x * 128 + (w & 3) * 32;
  v8f acc00 = {}, acc01 = {}, acc10 = {}, acc11 = {};
  const u16* Ab = Ctx + rowbase * HIDN;
  const u16* Bb = Wo + colbase * HIDN;
  for (int k = 0; k < HIDN; k += 32) {
    v16bf a0 = load_frag(Ab + k, HIDN);
    v16bf a1 = load_frag(Ab + 16 * HIDN + k, HIDN);
    v16bf b0 = load_frag(Bb + k, HIDN);
    v16bf b1 = load_frag(Bb + 16 * HIDN + k, HIDN);
    acc00 = wmma_bf16(a0, b0, acc00);
    acc01 = wmma_bf16(a0, b1, acc01);
    acc10 = wmma_bf16(a1, b0, acc10);
    acc11 = wmma_bf16(a1, b1, acc11);
  }
#pragma unroll
  for (int i = 0; i < 2; ++i) {
#pragma unroll
    for (int j = 0; j < 2; ++j) {
      const v8f& acc = (i == 0) ? (j == 0 ? acc00 : acc01) : (j == 0 ? acc10 : acc11);
      int col = colbase + j * 16 + (lane & 15);
      float b = bo[col];
      int r0 = rowbase + i * 16 + ((lane >> 4) << 3);
#pragma unroll
      for (int e = 0; e < 8; ++e) Out[(r0 + e) * HIDN + col] = acc[e] + b;
    }
  }
}

extern "C" void kernel_launch(void* const* d_in, const int* in_sizes, int n_in,
                              void* d_out, int out_size, void* d_ws, size_t ws_size,
                              hipStream_t stream) {
  (void)in_sizes; (void)n_in; (void)out_size; (void)ws_size;
  const float* hs   = (const float*)d_in[0];
  const float* mask = (const float*)d_in[1];
  const float* Wq   = (const float*)d_in[2];
  const float* bq   = (const float*)d_in[3];
  const float* Wk   = (const float*)d_in[4];
  const float* bk   = (const float*)d_in[5];
  const float* Wv   = (const float*)d_in[6];
  const float* bv   = (const float*)d_in[7];
  const float* Wo   = (const float*)d_in[8];
  const float* bo   = (const float*)d_in[9];
  const float* om   = (const float*)d_in[10];
  const float* rffb = (const float*)d_in[11];
  float* out = (float*)d_out;

  char* p = (char*)d_ws;
  auto carve = [&](size_t bytes) { void* r = (void*)p; p += (bytes + 255) & ~(size_t)255; return r; };
  u16* Xbf  = (u16*)carve((size_t)SEQ * HIDN * 2);
  u16* Wqb  = (u16*)carve((size_t)HIDN * HIDN * 2);
  u16* Wkb  = (u16*)carve((size_t)HIDN * HIDN * 2);
  u16* Wvb  = (u16*)carve((size_t)HIDN * HIDN * 2);
  u16* Wob  = (u16*)carve((size_t)HIDN * HIDN * 2);
  u16* Qh   = (u16*)carve((size_t)NH * SEQ * HD * 2);
  u16* Kh   = (u16*)carve((size_t)NH * SEQ * HD * 2);
  u16* Vt   = (u16*)carve((size_t)NH * SEQ * HD * 2);
  u16* Qn   = (u16*)carve((size_t)NH * SEQ * HD * 2);
  u16* Kn   = (u16*)carve((size_t)NH * SEQ * HD * 2);
  u16* omT  = (u16*)carve((size_t)NH * NM * HD * 2);
  u16* PhiQ = (u16*)carve((size_t)NH * SEQ * NM * 2);
  u16* PhiK = (u16*)carve((size_t)NH * SEQ * NM * 2);
  u16* Ctx  = (u16*)carve((size_t)SEQ * HIDN * 2);

  int nX = SEQ * HIDN, nW = HIDN * HIDN;
  k_conv_f32_bf16<<<(nX + 255) / 256, 256, 0, stream>>>(hs, Xbf, nX);
  k_conv_f32_bf16<<<(nW + 255) / 256, 256, 0, stream>>>(Wq, Wqb, nW);
  k_conv_f32_bf16<<<(nW + 255) / 256, 256, 0, stream>>>(Wk, Wkb, nW);
  k_conv_f32_bf16<<<(nW + 255) / 256, 256, 0, stream>>>(Wv, Wvb, nW);
  k_conv_f32_bf16<<<(nW + 255) / 256, 256, 0, stream>>>(Wo, Wob, nW);
  k_omega_t<<<(NH * HD * NM + 255) / 256, 256, 0, stream>>>(om, omT);

  k_qkv<<<dim3(HIDN / 128, SEQ / 64, 3), 256, 0, stream>>>(Xbf, Wqb, Wkb, Wvb, bq, bk, bv, Qh, Kh, Vt);
  k_norm<<<(NH * SEQ + 255) / 256, 256, 0, stream>>>(Qh, Qn);
  k_norm<<<(NH * SEQ + 255) / 256, 256, 0, stream>>>(Kh, Kn);
  k_phi<<<dim3(SEQ / 16, NH, 2), 256, 0, stream>>>(Qn, Kn, omT, rffb, PhiQ, PhiK);
  k_attn<<<dim3(SEQ / 16, NH), 256, 0, stream>>>(Qh, Kh, Vt, PhiQ, PhiK, mask, Ctx);
  k_out<<<dim3(HIDN / 128, SEQ / 64), 256, 0, stream>>>(Ctx, Wob, bo, out);
}
